// WeekForecastProbInformer_1151051235842
// MI455X (gfx1250) — compile-verified, matrix operand order checked by reading, not run on hardware
//
#include <hip/hip_runtime.h>
#include <math.h>

// ---------------- model dims ----------------
#define NBATCH 64
#define DM     256
#define NH     8
#define HDIM   32
#define DFF    512
#define LE0    336
#define LE2    84
#define LDEC   192
#define LABELL 24
#define PREDL  168
#define CHUNK  8           // batches per attention chunk

typedef __attribute__((ext_vector_type(16))) __bf16 v16bf;
typedef __attribute__((ext_vector_type(8)))  float  v8f;
typedef __attribute__((ext_vector_type(8)))  unsigned short ush8;

__device__ __forceinline__ unsigned short f2bf(float f) {
  unsigned int u = __float_as_uint(f);
  u += 0x7fffu + ((u >> 16) & 1u);           // round-to-nearest-even
  return (unsigned short)(u >> 16);
}
// generic (shared) pointer -> 32-bit LDS byte offset for async-to-LDS VDST
__device__ __forceinline__ unsigned lds_addr(const void* p) {
  return (unsigned)(unsigned long long)(const __attribute__((address_space(3))) void*)p;
}

// ---------------- WMMA GEMM: C = act(alpha*A@B^T + bias [+C]) ----------------
// A: bf16 [M,K] lda ; B: bf16 [N,K] ldb (always "transposed" layout) ;
// C: f32 [M,N] ldc.  REQUIREMENT: K % 32 == 0 and both operands zero-padded in K.
// Batched via z: off = (z/hc)*Outer + (z%hc)*Inner for A/B/C.
// 64x64 block tile, 8 waves (4 my x 2 nx), each wave = 16x32 (2 wmma / K-step).
// Interior tiles stage LDS via GLOBAL_LOAD_ASYNC_TO_LDS_B128 (ASYNCcnt path);
// edge tiles use guarded b128 loads with zero fill.
__global__ __launch_bounds__(256) void k_gemm(
    const unsigned short* __restrict__ A, const unsigned short* __restrict__ B,
    float* __restrict__ C, const float* __restrict__ bias,
    int M, int N, int K, int lda, int ldb, int ldc,
    float alpha, int accum, int act, int hc,
    long aO, long aI, long bO, long bI, long cO, long cI)
{
  __shared__ unsigned short As[64][40];   // 80B row stride -> 16B aligned
  __shared__ unsigned short Bs[64][40];
  const int z = blockIdx.z;
  const unsigned short* Ab = A + (long)(z / hc) * aO + (long)(z % hc) * aI;
  const unsigned short* Bb = B + (long)(z / hc) * bO + (long)(z % hc) * bI;
  float* Cb = C + (long)(z / hc) * cO + (long)(z % hc) * cI;

  const int tid = threadIdx.x;
  const int rowBase = blockIdx.y * 64, colBase = blockIdx.x * 64;
  const int wid = tid >> 5, lane = tid & 31;
  const int my = wid >> 1, nx = wid & 1;
  const int half = lane >> 4, mm = lane & 15;
  const int lr = tid >> 2;     // 0..63 : row (A) / col (B) for coop load
  const int kg = tid & 3;      // 0..3  : 8-element k-group
  const bool full = (rowBase + 64 <= M) && (colBase + 64 <= N);  // block-uniform

  const unsigned la = lds_addr(&As[lr][kg * 8]);
  const unsigned lb = lds_addr(&Bs[lr][kg * 8]);

  v8f acc0 = {}, acc1 = {};
  for (int kk = 0; kk < K; kk += 32) {
    if (full) {
      // async DMA into LDS: no VGPR round-trip, tracked by ASYNCcnt
      unsigned long long ga =
          (unsigned long long)(Ab + (long)(rowBase + lr) * lda + kk + kg * 8);
      unsigned long long gb =
          (unsigned long long)(Bb + (long)(colBase + lr) * ldb + kk + kg * 8);
      asm volatile("global_load_async_to_lds_b128 %0, %1, off"
                   :: "v"(la), "v"(ga) : "memory");
      asm volatile("global_load_async_to_lds_b128 %0, %1, off"
                   :: "v"(lb), "v"(gb) : "memory");
      asm volatile("s_wait_asynccnt 0x0" ::: "memory");
    } else {
      int gr = rowBase + lr;
      ush8 va = {};
      if (gr < M) va = *(const ush8*)(Ab + (long)gr * lda + kk + kg * 8);
      *(ush8*)&As[lr][kg * 8] = va;
      int gc = colBase + lr;
      ush8 vb = {};
      if (gc < N) vb = *(const ush8*)(Bb + (long)gc * ldb + kk + kg * 8);
      *(ush8*)&Bs[lr][kg * 8] = vb;
    }
    if (kk + 32 < K) {  // speculative prefetch of next A tile -> global_prefetch_b8
      int pr = rowBase + lr; if (pr >= M) pr = M - 1;
      __builtin_prefetch(Ab + (long)pr * lda + kk + 32, 0, 1);
    }
    __syncthreads();
    union { v16bf v; ush8 h[2]; } af, bf0, bf1;
    af.h[0]  = *(const ush8*)&As[(my << 4) + mm][half << 3];
    af.h[1]  = *(const ush8*)&As[(my << 4) + mm][16 + (half << 3)];
    bf0.h[0] = *(const ush8*)&Bs[(nx << 5) + mm][half << 3];
    bf0.h[1] = *(const ush8*)&Bs[(nx << 5) + mm][16 + (half << 3)];
    bf1.h[0] = *(const ush8*)&Bs[(nx << 5) + 16 + mm][half << 3];
    bf1.h[1] = *(const ush8*)&Bs[(nx << 5) + 16 + mm][16 + (half << 3)];
    acc0 = __builtin_amdgcn_wmma_f32_16x16x32_bf16(false, af.v, false, bf0.v,
                                                   (short)0, acc0, false, false);
    acc1 = __builtin_amdgcn_wmma_f32_16x16x32_bf16(false, af.v, false, bf1.v,
                                                   (short)0, acc1, false, false);
    __syncthreads();
  }
  #pragma unroll
  for (int t = 0; t < 2; t++) {
    v8f acc = t ? acc1 : acc0;
    int col = colBase + (nx << 5) + (t << 4) + mm;
    #pragma unroll
    for (int r = 0; r < 8; r++) {
      int row = rowBase + (my << 4) + (half << 3) + r;  // ISA C layout: m = 8*half + r
      if (row < M && col < N) {
        float v = alpha * acc[r];
        if (bias) v += bias[col];
        long off = (long)row * ldc + col;
        if (accum) v += Cb[off];
        if (act == 1) v = 0.5f * v * (1.0f + erff(v * 0.70710678118f));  // exact GELU
        Cb[off] = v;
      }
    }
  }
}

// ---------------- elementwise / small kernels ----------------
__global__ void k_cvt(const float* __restrict__ x, unsigned short* __restrict__ y, long n) {
  long i = (long)blockIdx.x * blockDim.x + threadIdx.x;
  if (i < n) y[i] = f2bf(x[i]);
}
// weights (K,N) f32 -> bf16 [N][K]
__global__ void k_cvtwt(const float* __restrict__ w, unsigned short* __restrict__ o,
                        int K, int N) {
  int i = blockIdx.x * blockDim.x + threadIdx.x;
  if (i >= K * N) return;
  int n = i / K, k = i % K;
  o[i] = f2bf(w[(long)k * N + n]);
}
// V (b,Lkv,h,32) f32 -> bf16 [b][h][d][Lkp] (zero-padded s)
__global__ void k_cvtv(const float* __restrict__ V, unsigned short* __restrict__ o,
                       int Lkv, int Lkp) {
  long i = (long)blockIdx.x * blockDim.x + threadIdx.x;
  long tot = (long)NBATCH * NH * HDIM * Lkp;
  if (i >= tot) return;
  int s = (int)(i % Lkp); long r = i / Lkp;
  int d = (int)(r % HDIM); long bh = r / HDIM;
  int b = (int)(bh / NH), h = (int)(bh % NH);
  o[i] = (s < Lkv) ? f2bf(V[((long)b * Lkv + s) * DM + h * HDIM + d]) : (unsigned short)0;
}
__global__ void k_addip(float* __restrict__ y, const float* __restrict__ x, long n) {
  long i = (long)blockIdx.x * blockDim.x + threadIdx.x;
  if (i < n) y[i] += x[i];
}
__global__ void k_buildxe(const float* a, const float* b, const float* c, const float* d,
                          float* xe) {
  int i = blockIdx.x * blockDim.x + threadIdx.x;
  if (i >= NBATCH * LE0 * 4) return;
  int bb = i / (LE0 * 4), r = i % (LE0 * 4), t = r / 4, ch = r % 4;
  const float* src = (ch == 0) ? a : (ch == 1) ? b : (ch == 2) ? c : d;
  xe[i] = src[bb * LE0 + t];
}
__global__ void k_buildxd(const float* xe, float* xd) {
  int i = blockIdx.x * blockDim.x + threadIdx.x;
  if (i >= NBATCH * LDEC * 4) return;
  int bb = i / (LDEC * 4), r = i % (LDEC * 4), t = r / 4, ch = r % 4;
  xd[i] = (t < LABELL) ? xe[(bb * LE0 + (LE0 - LABELL) + t) * 4 + ch] : 0.0f;
}
// y[b,t,o] = circ-conv3(x(..,4), w(256,4,3)) + pos_emb(t,o) + temp_emb(o)
__global__ void k_embed(const float* __restrict__ x, const float* __restrict__ w,
                        float* __restrict__ y, int L) {
  int i = blockIdx.x * blockDim.x + threadIdx.x;
  if (i >= NBATCH * L * DM) return;
  int bb = i / (L * DM), r = i % (L * DM), t = r / DM, o = r % DM;
  float s = 0.f;
  #pragma unroll
  for (int k = 0; k < 3; k++) {
    int tt = t + k - 1; tt = (tt < 0) ? tt + L : (tt >= L) ? tt - L : tt;
    const float* xr = x + (long)(bb * L + tt) * 4;
    #pragma unroll
    for (int c = 0; c < 4; c++) s += xr[c] * w[(o * 4 + c) * 3 + k];
  }
  float div = __expf(-logf(10000.f) * (float)(o & ~1) / (float)DM);
  float ang = (float)t * div;
  s += (o & 1) ? __cosf(ang) : __sinf(ang);
  s += (o & 1) ? 4.0f : 0.0f;
  y[i] = s;
}
__global__ __launch_bounds__(256) void k_ln256(const float* __restrict__ X, float* __restrict__ Y,
                                               const float* __restrict__ g, const float* __restrict__ b) {
  __shared__ float sm[256];
  int row = blockIdx.x, t = threadIdx.x;
  float v = X[(long)row * DM + t];
  sm[t] = v; __syncthreads();
  for (int s = 128; s > 0; s >>= 1) { if (t < s) sm[t] += sm[t + s]; __syncthreads(); }
  float mean = sm[0] / (float)DM; __syncthreads();
  float d = v - mean;
  sm[t] = d * d; __syncthreads();
  for (int s = 128; s > 0; s >>= 1) { if (t < s) sm[t] += sm[t + s]; __syncthreads(); }
  float var = sm[0] / (float)DM;
  Y[(long)row * DM + t] = d * rsqrtf(var + 1e-5f) * g[t] + b[t];
}
// row softmax (optionally causal) on f32 scores; writes bf16 probs, stride Lkp, zero-padded
__global__ void k_softmax(const float* __restrict__ S, unsigned short* __restrict__ P,
                          int Lq, int Lk, int Lkp, int causal) {
  int row = blockIdx.x, lane = threadIdx.x;
  int l = row % Lq;
  const float* sr = S + (long)row * Lk;
  unsigned short* pr = P + (long)row * Lkp;
  int lim = causal ? (l + 1) : Lk;
  float mx = -3.4e38f;
  for (int c = lane; c < lim; c += 32) mx = fmaxf(mx, sr[c]);
  for (int o = 16; o > 0; o >>= 1) mx = fmaxf(mx, __shfl_xor(mx, o));
  float sum = 0.f;
  for (int c = lane; c < lim; c += 32) sum += __expf(sr[c] - mx);
  for (int o = 16; o > 0; o >>= 1) sum += __shfl_xor(sum, o);
  float inv = 1.f / sum;
  for (int c = lane; c < Lkp; c += 32)
    pr[c] = f2bf((c < lim) ? __expf(sr[c] - mx) * inv : 0.f);
}
__global__ void k_padcopy(const float* __restrict__ X, float* __restrict__ XP, int L) {
  int i = blockIdx.x * blockDim.x + threadIdx.x;
  if (i >= NBATCH * (L + 2) * DM) return;
  int bb = i / ((L + 2) * DM), r = i % ((L + 2) * DM), row = r / DM, o = r % DM;
  int src = (row == 0) ? (L - 1) : (row == L + 1) ? 0 : (row - 1);
  XP[i] = X[((long)bb * L + src) * DM + o];
}
// (256,256,3) -> 3 packed [N=o][K=i] bf16 matrices: out[tap][o*256+i] = w[o,i,tap]
__global__ void k_pack_distil(const float* __restrict__ w, unsigned short* __restrict__ out) {
  int i = blockIdx.x * blockDim.x + threadIdx.x;
  if (i >= 3 * 256 * 256) return;
  int k = i / 65536, rem = i % 65536, o = rem / 256, ii = rem % 256;
  out[i] = f2bf(w[(o * 256 + ii) * 3 + k]);
}
__global__ void k_bnelu(float* __restrict__ y, const float* __restrict__ g,
                        const float* __restrict__ b, long n) {
  long i = (long)blockIdx.x * blockDim.x + threadIdx.x;
  if (i >= n) return;
  int o = (int)(i % DM);
  float v = y[i] * 0.99999500003750f * g[o] + b[o];
  y[i] = (v > 0.f) ? v : expm1f(v);
}
__global__ void k_maxpool(const float* __restrict__ X, float* __restrict__ Y, int L, int Lo) {
  int i = blockIdx.x * blockDim.x + threadIdx.x;
  if (i >= NBATCH * Lo * DM) return;
  int bb = i / (Lo * DM), r = i % (Lo * DM), t = r / DM, o = r % DM;
  float m = -3.4e38f;
  for (int k = -1; k <= 1; k++) {
    int tt = 2 * t + k;
    if (tt >= 0 && tt < L) m = fmaxf(m, X[((long)bb * L + tt) * DM + o]);
  }
  Y[i] = m;
}
__global__ void k_proj(const float* __restrict__ D, const float* __restrict__ w,
                       const float* __restrict__ b, float* __restrict__ out) {
  int i = blockIdx.x * blockDim.x + threadIdx.x;
  if (i >= 2 * NBATCH * PREDL) return;
  int c = i / (NBATCH * PREDL), rem = i % (NBATCH * PREDL), bb = rem / PREDL, t = rem % PREDL;
  const float* dr = D + ((long)bb * LDEC + LABELL + t) * DM;
  float s = b[c];
  for (int k = 0; k < DM; k++) s += dr[k] * w[k * 2 + c];
  out[i] = s;  // [out0 flat | out1 flat]
}

// ---------------- host-side plumbing ----------------
struct WS {
  float *XE, *XD, *X, *T1, *Qf, *Kf, *Vf, *CTX, *FF, *XP, *D, *ENC, *Sb;
  unsigned short *AB, *QB, *KB, *VB, *PB, *WB;
};
static WS carve(void* ws) {
  char* p = (char*)ws; size_t o = 0; WS w;
  auto F = [&](size_t n) { float* r = (float*)(p + o); o += ((n * 4 + 255) & ~(size_t)255); return r; };
  auto H = [&](size_t n) { unsigned short* r = (unsigned short*)(p + o); o += ((n * 2 + 255) & ~(size_t)255); return r; };
  w.XE = F((size_t)NBATCH * LE0 * 4);
  w.XD = F((size_t)NBATCH * LDEC * 4);
  w.X  = F((size_t)NBATCH * LE0 * DM);
  w.T1 = F((size_t)NBATCH * LE0 * DM);
  w.Qf = F((size_t)NBATCH * LE0 * DM);
  w.Kf = F((size_t)NBATCH * LE0 * DM);
  w.Vf = F((size_t)NBATCH * LE0 * DM);
  w.CTX= F((size_t)NBATCH * LE0 * DM);
  w.FF = F((size_t)NBATCH * LE0 * DFF);
  w.XP = F((size_t)NBATCH * (LE0 + 2) * DM);
  w.D  = F((size_t)NBATCH * LDEC * DM);
  w.ENC= F((size_t)NBATCH * LE2 * DM);
  w.Sb = F((size_t)CHUNK * NH * LE0 * LE0);
  w.AB = H((size_t)NBATCH * LE0 * DFF);
  w.QB = H((size_t)NBATCH * LE0 * DM);
  w.KB = H((size_t)NBATCH * LE0 * DM);
  w.VB = H((size_t)NBATCH * LE0 * DM);       // holds [b][h][d][Lkp] transposed V
  w.PB = H((size_t)CHUNK * NH * LE0 * LE0);  // padded-stride probs
  w.WB = H((size_t)3 * 256 * 256);
  return w;
}
static inline void gemm(hipStream_t st, const unsigned short* A, const unsigned short* B,
                        float* C, const float* bias, int M, int N, int K,
                        int lda, int ldb, int ldc, float al, int acc, int act,
                        int nb = 1, int hc = 1, long aO = 0, long aI = 0,
                        long bO = 0, long bI = 0, long cO = 0, long cI = 0) {
  dim3 g((N + 63) / 64, (M + 63) / 64, nb);
  k_gemm<<<g, dim3(256), 0, st>>>(A, B, C, bias, M, N, K, lda, ldb, ldc,
                                  al, acc, act, hc, aO, aI, bO, bI, cO, cI);
}
static inline void cvt(hipStream_t st, const float* x, unsigned short* y, long n) {
  k_cvt<<<(unsigned)((n + 255) / 256), 256, 0, st>>>(x, y, n);
}
static inline void cvtwt(hipStream_t st, const float* x, unsigned short* y, int K, int N) {
  k_cvtwt<<<(K * N + 255) / 256, 256, 0, st>>>(x, y, K, N);
}
static inline void addip(hipStream_t st, float* y, const float* x, long n) {
  k_addip<<<(unsigned)((n + 255) / 256), 256, 0, st>>>(y, x, n);
}
static inline void ln(hipStream_t st, const float* x, float* y, const float* g,
                      const float* b, int rows) {
  k_ln256<<<rows, 256, 0, st>>>(x, y, g, b);
}

struct AT { const float *qw,*qb,*kw,*kb,*vw,*vb,*ow,*ob; };
struct EL { AT a; const float *c1w,*c1b,*c2w,*c2b,*n1g,*n1b,*n2g,*n2b; };
struct DS { const float *w,*b,*g,*bb; };
struct DL { AT s, c; const float *c1w,*c1b,*c2w,*c2b,*n1g,*n1b,*n2g,*n2b,*n3g,*n3b; };

// dense-analog attention (full softmax; causal for decoder self; mix = bhld output)
static void run_attn(hipStream_t st, const WS& w, const float* xq, int Lq,
                     const float* xkv, int Lkv, const AT& p, float* out,
                     int causal, int mix) {
  const float SC = 0.17677669529663687f;  // 1/sqrt(32)
  const int Lkp = (Lkv + 31) & ~31;
  int Mq = NBATCH * Lq, Mk = NBATCH * Lkv;
  cvt(st, xq, w.AB, (long)Mq * DM);
  cvtwt(st, p.qw, w.WB, DM, DM);
  gemm(st, w.AB, w.WB, w.Qf, p.qb, Mq, DM, DM, DM, DM, DM, 1.f, 0, 0);
  cvt(st, xkv, w.AB, (long)Mk * DM);
  cvtwt(st, p.kw, w.WB, DM, DM);
  gemm(st, w.AB, w.WB, w.Kf, p.kb, Mk, DM, DM, DM, DM, DM, 1.f, 0, 0);
  cvtwt(st, p.vw, w.WB, DM, DM);
  gemm(st, w.AB, w.WB, w.Vf, p.vb, Mk, DM, DM, DM, DM, DM, 1.f, 0, 0);
  cvt(st, w.Qf, w.QB, (long)Mq * DM);
  cvt(st, w.Kf, w.KB, (long)Mk * DM);
  { long n = (long)NBATCH * NH * HDIM * Lkp;
    k_cvtv<<<(unsigned)((n + 255) / 256), 256, 0, st>>>(w.Vf, w.VB, Lkv, Lkp); }
  for (int cb = 0; cb < NBATCH / CHUNK; cb++) {
    const unsigned short* Qc = w.QB + (size_t)cb * CHUNK * Lq * DM;
    const unsigned short* Kc = w.KB + (size_t)cb * CHUNK * Lkv * DM;
    const unsigned short* Vc = w.VB + (size_t)cb * CHUNK * NH * HDIM * Lkp;
    // scores[bh,l,s] = Q @ K^T * SC   (B = K in [s][d] = [N][K] layout)
    gemm(st, Qc, Kc, w.Sb, nullptr, Lq, Lkv, HDIM, DM, DM, Lkv, SC, 0, 0,
         CHUNK * NH, NH, (long)Lq * DM, HDIM, (long)Lkv * DM, HDIM,
         (long)NH * Lq * Lkv, (long)Lq * Lkv);
    k_softmax<<<CHUNK * NH * Lq, 32, 0, st>>>(w.Sb, w.PB, Lq, Lkv, Lkp, causal);
    float* Cc; long cO, cI; int ldc;
    if (mix) { Cc = w.CTX + (size_t)cb * CHUNK * NH * Lq * HDIM;
               cO = (long)NH * Lq * HDIM; cI = (long)Lq * HDIM; ldc = HDIM; }
    else     { Cc = w.CTX + (size_t)cb * CHUNK * Lq * DM;
               cO = (long)Lq * DM; cI = HDIM; ldc = DM; }
    // ctx = P @ V  (B = V^T in [d][s] layout, K = padded Lkp, zero tails)
    gemm(st, w.PB, Vc, Cc, nullptr, Lq, HDIM, Lkp, Lkp, Lkp, ldc, 1.f, 0, 0,
         CHUNK * NH, NH, (long)NH * Lq * Lkp, (long)Lq * Lkp,
         (long)NH * HDIM * Lkp, (long)HDIM * Lkp, cO, cI);
  }
  cvt(st, w.CTX, w.AB, (long)Mq * DM);
  cvtwt(st, p.ow, w.WB, DM, DM);
  gemm(st, w.AB, w.WB, out, p.ob, Mq, DM, DM, DM, DM, DM, 1.f, 0, 0);
}
static void run_ff(hipStream_t st, const WS& w, float* x, int L,
                   const float* c1w, const float* c1b, const float* c2w, const float* c2b) {
  int M = NBATCH * L;
  cvt(st, x, w.AB, (long)M * DM);
  cvtwt(st, c1w, w.WB, DM, DFF);
  gemm(st, w.AB, w.WB, w.FF, c1b, M, DFF, DM, DM, DM, DFF, 1.f, 0, 1);
  cvt(st, w.FF, w.AB, (long)M * DFF);
  cvtwt(st, c2w, w.WB, DFF, DM);
  gemm(st, w.AB, w.WB, w.T1, c2b, M, DM, DFF, DFF, DFF, DM, 1.f, 0, 0);
  addip(st, x, w.T1, (long)M * DM);
}

extern "C" void kernel_launch(void* const* d_in, const int* in_sizes, int n_in,
                              void* d_out, int out_size, void* d_ws, size_t ws_size,
                              hipStream_t stream) {
  (void)in_sizes; (void)n_in; (void)out_size; (void)ws_size;
  WS w = carve(d_ws);
  const float* enc_l = (const float*)d_in[0];
  const float* enc_t = (const float*)d_in[1];
  const float* enc_w = (const float*)d_in[2];
  const float* enc_s = (const float*)d_in[3];
  int pi = 4;
  auto P = [&]() { return (const float*)d_in[pi++]; };
  const float* enc_emb = P();
  const float* dec_emb = P();
  EL enc[3];
  for (int i = 0; i < 3; i++) {
    enc[i].a.qw = P(); enc[i].a.qb = P(); enc[i].a.kw = P(); enc[i].a.kb = P();
    enc[i].a.vw = P(); enc[i].a.vb = P(); enc[i].a.ow = P(); enc[i].a.ob = P();
    enc[i].c1w = P(); enc[i].c1b = P(); enc[i].c2w = P(); enc[i].c2b = P();
    enc[i].n1g = P(); enc[i].n1b = P(); enc[i].n2g = P(); enc[i].n2b = P();
  }
  DS ds[2];
  for (int i = 0; i < 2; i++) { ds[i].w = P(); ds[i].b = P(); ds[i].g = P(); ds[i].bb = P(); }
  const float* engc = P(); const float* enbc = P();
  DL dec[2];
  for (int i = 0; i < 2; i++) {
    AT* as[2] = { &dec[i].s, &dec[i].c };
    for (int j = 0; j < 2; j++) {
      as[j]->qw = P(); as[j]->qb = P(); as[j]->kw = P(); as[j]->kb = P();
      as[j]->vw = P(); as[j]->vb = P(); as[j]->ow = P(); as[j]->ob = P();
    }
    dec[i].c1w = P(); dec[i].c1b = P(); dec[i].c2w = P(); dec[i].c2b = P();
    dec[i].n1g = P(); dec[i].n1b = P(); dec[i].n2g = P(); dec[i].n2b = P();
    dec[i].n3g = P(); dec[i].n3b = P();
  }
  const float* dngc = P(); const float* dnbc = P();
  const float* projw = P(); const float* projb = P();

  // inputs -> x_enc / x_dec
  k_buildxe<<<(NBATCH * LE0 * 4 + 255) / 256, 256, 0, stream>>>(enc_l, enc_t, enc_w, enc_s, w.XE);
  k_buildxd<<<(NBATCH * LDEC * 4 + 255) / 256, 256, 0, stream>>>(w.XE, w.XD);

  // ---------- encoder ----------
  k_embed<<<(NBATCH * LE0 * DM + 255) / 256, 256, 0, stream>>>(w.XE, enc_emb, w.X, LE0);
  int L = LE0;
  for (int i = 0; i < 3; i++) {
    run_attn(stream, w, w.X, L, w.X, L, enc[i].a, w.T1, 0, 0);
    addip(stream, w.X, w.T1, (long)NBATCH * L * DM);
    ln(stream, w.X, w.X, enc[i].n1g, enc[i].n1b, NBATCH * L);
    run_ff(stream, w, w.X, L, enc[i].c1w, enc[i].c1b, enc[i].c2w, enc[i].c2b);
    ln(stream, w.X, w.X, enc[i].n2g, enc[i].n2b, NBATCH * L);
    if (i < 2) {  // distil: circular-pad + 3 accumulated WMMA GEMMs + BN/ELU + maxpool
      k_padcopy<<<(NBATCH * (L + 2) * DM + 255) / 256, 256, 0, stream>>>(w.X, w.XP, L);
      cvt(stream, w.XP, w.AB, (long)NBATCH * (L + 2) * DM);
      k_pack_distil<<<(3 * 256 * 256 + 255) / 256, 256, 0, stream>>>(ds[i].w, w.WB);
      for (int k = 0; k < 3; k++)
        gemm(stream, w.AB + (size_t)k * DM, w.WB + (size_t)k * DM * DM, w.T1,
             (k == 0) ? ds[i].b : nullptr, L, DM, DM, DM, DM, DM, 1.f,
             (k == 0) ? 0 : 1, 0, NBATCH, 1,
             (long)(L + 2) * DM, 0, 0, 0, (long)L * DM, 0);
      k_bnelu<<<(unsigned)(((long)NBATCH * L * DM + 255) / 256), 256, 0, stream>>>(
          w.T1, ds[i].g, ds[i].bb, (long)NBATCH * L * DM);
      int Lo = (L - 1) / 2 + 1;
      k_maxpool<<<(NBATCH * Lo * DM + 255) / 256, 256, 0, stream>>>(w.T1, w.X, L, Lo);
      L = Lo;
    }
  }
  ln(stream, w.X, w.ENC, engc, enbc, NBATCH * L);   // L == 84

  // ---------- decoder ----------
  k_embed<<<(NBATCH * LDEC * DM + 255) / 256, 256, 0, stream>>>(w.XD, dec_emb, w.D, LDEC);
  for (int i = 0; i < 2; i++) {
    run_attn(stream, w, w.D, LDEC, w.D, LDEC, dec[i].s, w.T1, 1, 1);   // causal, mix
    addip(stream, w.D, w.T1, (long)NBATCH * LDEC * DM);
    ln(stream, w.D, w.D, dec[i].n1g, dec[i].n1b, NBATCH * LDEC);
    run_attn(stream, w, w.D, LDEC, w.ENC, LE2, dec[i].c, w.T1, 0, 0);  // cross
    addip(stream, w.D, w.T1, (long)NBATCH * LDEC * DM);
    ln(stream, w.D, w.D, dec[i].n2g, dec[i].n2b, NBATCH * LDEC);
    run_ff(stream, w, w.D, LDEC, dec[i].c1w, dec[i].c1b, dec[i].c2w, dec[i].c2b);
    ln(stream, w.D, w.D, dec[i].n3g, dec[i].n3b, NBATCH * LDEC);
  }
  ln(stream, w.D, w.D, dngc, dnbc, NBATCH * LDEC);
  k_proj<<<(2 * NBATCH * PREDL + 255) / 256, 256, 0, stream>>>(w.D, projw, projb, (float*)d_out);
}
